// BYOLLoss_24215025615474
// MI455X (gfx1250) — matrix operand order repaired and checked by reading, not compiled, not run on hardware
//
#include <hip/hip_runtime.h>
#include <hip/hip_bf16.h>

// ---------------------------------------------------------------------------
// Problem constants (from reference setup_inputs)
// ---------------------------------------------------------------------------
#define BSZ   4096            // batch per view
#define N2    8192            // 2*B
#define DFE   2048            // backbone feature dim
#define DP    256             // predictor dim
#define NCLS  1000            // classes
#define EPSF  1e-12f

typedef __attribute__((ext_vector_type(16))) __bf16 v16bf;
typedef __attribute__((ext_vector_type(8)))  __bf16 v8bf;
typedef __attribute__((ext_vector_type(8)))  float  v8f;

// ---------------------------------------------------------------------------
// CDNA5 async-to-LDS helpers (ASYNCcnt path; inline asm is the portable route)
// Generic LDS pointer low 32 bits == wave-relative LDS byte offset (ISA 10.2).
// ---------------------------------------------------------------------------
__device__ inline void async_copy_b128(const void* lds_dst, const void* gsrc) {
    uint32_t loff = (uint32_t)(uintptr_t)lds_dst;
    asm volatile("global_load_async_to_lds_b128 %0, %1, off"
                 :: "v"(loff), "v"(gsrc) : "memory");
}
__device__ inline void wait_async_le4() {
    asm volatile("s_wait_asynccnt 4" ::: "memory");
}
__device__ inline void wait_async_0() {
    asm volatile("s_wait_asynccnt 0" ::: "memory");
}

// ---------------------------------------------------------------------------
// Block reduction helpers (blockDim.x == 256 everywhere)
// ---------------------------------------------------------------------------
__device__ inline float blockSum256(float v, float* sf) {
    int t = threadIdx.x;
    sf[t] = v; __syncthreads();
    for (int s = 128; s > 0; s >>= 1) {
        if (t < s) sf[t] += sf[t + s];
        __syncthreads();
    }
    float r = sf[0]; __syncthreads();
    return r;
}

__device__ inline float blockMax256(float v, float* sf) {
    int t = threadIdx.x;
    sf[t] = v; __syncthreads();
    for (int s = 128; s > 0; s >>= 1) {
        if (t < s) sf[t] = fmaxf(sf[t], sf[t + s]);
        __syncthreads();
    }
    float r = sf[0]; __syncthreads();
    return r;
}

// ---------------------------------------------------------------------------
// WMMA fragment builders (bf16 16x16x32, CDNA5 VGPR layouts).
// A 16x32 (MxK): lane L row M=L%16; elems 0..7 -> K=(L/16)*8+i, 8..15 -> +16.
// B 32x16 (KxN) with B = M2^T: lane L col N=L%16; elems 0..7 -> K=(L/16)*8+i, 8..15 -> +16.
// All pieces are 16B-aligned v8bf loads (safe for LDS ds_load_b128).
// ---------------------------------------------------------------------------
__device__ inline v16bf make_a_frag(const __bf16* rowp, int half) {
    v8bf lo = *(const v8bf*)(rowp + half * 8);
    v8bf hi = *(const v8bf*)(rowp + 16 + half * 8);
    v16bf a;
#pragma unroll
    for (int i = 0; i < 8; ++i) { a[i] = lo[i]; a[i + 8] = hi[i]; }
    return a;
}
__device__ inline v16bf make_b_frag(const __bf16* rowp, int half) {
    v8bf lo = *(const v8bf*)(rowp + half * 8);
    v8bf hi = *(const v8bf*)(rowp + 16 + half * 8);
    v16bf b;
#pragma unroll
    for (int i = 0; i < 8; ++i) { b[i] = lo[i]; b[i + 8] = hi[i]; }
    return b;
}

// ---------------------------------------------------------------------------
// K1: fe0/fe1 -> bf16 concat [N2 x DFE] + fp32 row squared norms (of the
//     bf16-rounded values so the bf16 Gram diagonal gives ~0 self-distance).
// ---------------------------------------------------------------------------
__global__ __launch_bounds__(256) void k_prep_fe(const float* __restrict__ fe0,
                                                 const float* __restrict__ fe1,
                                                 __bf16* __restrict__ feb,
                                                 float* __restrict__ fen) {
    __shared__ float sf[256];
    int r = blockIdx.x;
    const float* src = (r < BSZ) ? fe0 + (size_t)r * DFE
                                 : fe1 + (size_t)(r - BSZ) * DFE;
    float s = 0.0f;
    for (int i = threadIdx.x; i < DFE; i += 256) {
        __bf16 b = (__bf16)src[i];
        feb[(size_t)r * DFE + i] = b;
        float xb = (float)b;
        s += xb * xb;
    }
    s = blockSum256(s, sf);
    if (threadIdx.x == 0) fen[r] = s;
}

// ---------------------------------------------------------------------------
// K2: O = [l2norm(p0); l2norm(p1)] as bf16 [N2 x DP]
// ---------------------------------------------------------------------------
__global__ __launch_bounds__(256) void k_prep_p(const float* __restrict__ p0,
                                                const float* __restrict__ p1,
                                                __bf16* __restrict__ ob) {
    __shared__ float sf[256];
    int r = blockIdx.x, t = threadIdx.x;    // DP == blockDim.x == 256
    const float* src = (r < BSZ) ? p0 + (size_t)r * DP
                                 : p1 + (size_t)(r - BSZ) * DP;
    float x = src[t];
    float s2 = blockSum256(x * x, sf);
    float inv = 1.0f / fmaxf(sqrtf(s2), EPSF);
    ob[(size_t)r * DP + t] = (__bf16)(x * inv);
}

// ---------------------------------------------------------------------------
// K3: per-row BYOL loss partials: (2-2*cos(p0,z1)) + (2-2*cos(p1,z0))
// ---------------------------------------------------------------------------
__global__ __launch_bounds__(256) void k_byol(const float* __restrict__ p0,
                                              const float* __restrict__ p1,
                                              const float* __restrict__ z0,
                                              const float* __restrict__ z1,
                                              float* __restrict__ part) {
    __shared__ float sf[256];
    int r = blockIdx.x, t = threadIdx.x;
    float a = p0[(size_t)r * DP + t], b = z1[(size_t)r * DP + t];
    float c = p1[(size_t)r * DP + t], d = z0[(size_t)r * DP + t];
    float aa = blockSum256(a * a, sf);
    float bb = blockSum256(b * b, sf);
    float ab = blockSum256(a * b, sf);
    float cc = blockSum256(c * c, sf);
    float dd = blockSum256(d * d, sf);
    float cd = blockSum256(c * d, sf);
    if (t == 0) {
        float s1 = ab / (fmaxf(sqrtf(aa), EPSF) * fmaxf(sqrtf(bb), EPSF));
        float s2 = cd / (fmaxf(sqrtf(cc), EPSF) * fmaxf(sqrtf(dd), EPSF));
        part[r] = (2.0f - 2.0f * s1) + (2.0f - 2.0f * s2);
    }
}

// ---------------------------------------------------------------------------
// K4: per-row cross-entropy partials over concat(cls0, cls1)
// ---------------------------------------------------------------------------
__global__ __launch_bounds__(256) void k_ce(const float* __restrict__ cls0,
                                            const float* __restrict__ cls1,
                                            const int* __restrict__ labels,
                                            float* __restrict__ part) {
    __shared__ float sf[256];
    int r = blockIdx.x, t = threadIdx.x;
    const float* row = (r < BSZ) ? cls0 + (size_t)r * NCLS
                                 : cls1 + (size_t)(r - BSZ) * NCLS;
    int lab = labels[r & (BSZ - 1)];
    float mx = -3.4e38f;
    for (int i = t; i < NCLS; i += 256) mx = fmaxf(mx, row[i]);
    mx = blockMax256(mx, sf);
    float se = 0.0f;
    for (int i = t; i < NCLS; i += 256) se += __expf(row[i] - mx);
    se = blockSum256(se, sf);
    if (t == 0) part[r] = -(row[lab] - mx - logf(se));
}

// ---------------------------------------------------------------------------
// K5: mean-reduce partial array -> single output scalar
// ---------------------------------------------------------------------------
__global__ __launch_bounds__(256) void k_reduce_mean(const float* __restrict__ in,
                                                     int n, float scale,
                                                     float* __restrict__ out) {
    __shared__ float sf[256];
    float s = 0.0f;
    for (int i = threadIdx.x; i < n; i += 256) s += in[i];
    s = blockSum256(s, sf);
    if (threadIdx.x == 0) out[0] = s * scale;
}

// ---------------------------------------------------------------------------
// K6: labels2 = tile(arange(B), 2), emitted as float (d_out dtype)
// ---------------------------------------------------------------------------
__global__ __launch_bounds__(256) void k_labels(float* __restrict__ out) {
    int i = blockIdx.x * 256 + threadIdx.x;
    if (i < N2) out[i] = (float)(i & (BSZ - 1));
}

// ---------------------------------------------------------------------------
// K7: fe Gram via WMMA bf16 with async-to-LDS double-buffered panel staging.
//     Macro tile 128x128 per block, 8 waves each owning a 32x64 sub-grid
//     (2x4 16x16 WMMA tiles, 64 f32 acc VGPRs/lane).
//     Per k-step: A,B panels 128x32 bf16 staged by global_load_async_to_lds
//     (4 async instr / wave / panel -> s_wait_asynccnt 4 overlaps next panel).
// ---------------------------------------------------------------------------
#define GMT 128               // macro tile M
#define GNT 128               // macro tile N
#define GKT 32                // k step
#define LDP 40                // padded LDS row (bf16 elems): 80B, 16B-aligned

__global__ __launch_bounds__(256) void k_gram_fe(const __bf16* __restrict__ feb,
                                                 float* __restrict__ gram) {
    __shared__ __attribute__((aligned(64))) __bf16 sA[2][GMT * LDP];
    __shared__ __attribute__((aligned(64))) __bf16 sB[2][GNT * LDP];

    const int t    = threadIdx.x;
    const int lane = t & 31;
    const int wave = t >> 5;
    const int nbx  = N2 / GNT;                 // 64
    const int m0   = (blockIdx.x / nbx) * GMT;
    const int n0   = (blockIdx.x % nbx) * GNT;

    const int wm = wave >> 1;                  // 0..3 : 32-row strip
    const int wn = wave & 1;                   // 0..1 : 64-col strip
    const int half = lane >> 4;
    const int lrow = lane & 15;

    // panel copy: 512 x 16B chunks each for A and B; thread does chunks t,t+256
    auto issue_panel = [&](int buf, int kk) {
        for (int c = t; c < 512; c += 256) {
            int row = c >> 2;                  // 0..127
            int c16 = c & 3;                   // 16B chunk within 64B row-slice
            const __bf16* ga = feb + (size_t)(m0 + row) * DFE + kk + c16 * 8;
            const __bf16* gb = feb + (size_t)(n0 + row) * DFE + kk + c16 * 8;
            async_copy_b128(&sA[buf][row * LDP + c16 * 8], ga);
            async_copy_b128(&sB[buf][row * LDP + c16 * 8], gb);
        }
    };

    v8f acc[2][4] = {};
    issue_panel(0, 0);
    const int nsteps = DFE / GKT;              // 64
    for (int s = 0; s < nsteps; ++s) {
        int buf = s & 1;
        if (s + 1 < nsteps) {
            issue_panel(buf ^ 1, (s + 1) * GKT);
            wait_async_le4();                  // panel s landed (in-order)
        } else {
            wait_async_0();
        }
        __syncthreads();

        v16bf afrag[2], bfrag[4];
#pragma unroll
        for (int mt = 0; mt < 2; ++mt)
            afrag[mt] = make_a_frag(&sA[buf][(wm * 32 + mt * 16 + lrow) * LDP], half);
#pragma unroll
        for (int nt = 0; nt < 4; ++nt)
            bfrag[nt] = make_b_frag(&sB[buf][(wn * 64 + nt * 16 + lrow) * LDP], half);
#pragma unroll
        for (int mt = 0; mt < 2; ++mt)
#pragma unroll
            for (int nt = 0; nt < 4; ++nt)
                acc[mt][nt] = __builtin_amdgcn_wmma_f32_16x16x32_bf16(
                    false, afrag[mt], false, bfrag[nt], (short)0, acc[mt][nt],
                    false, false);
        __syncthreads();                       // all waves done reading buf
    }

    // store: C/D layout -> VGPR v holds (M = v + 8*half, N = lane%16)
#pragma unroll
    for (int mt = 0; mt < 2; ++mt) {
        int mbase = m0 + wm * 32 + mt * 16 + half * 8;
#pragma unroll
        for (int nt = 0; nt < 4; ++nt) {
            int col = n0 + wn * 64 + nt * 16 + lrow;
#pragma unroll
            for (int v = 0; v < 8; ++v)
                gram[(size_t)(mbase + v) * N2 + col] = acc[mt][nt][v];
        }
    }
}

// ---------------------------------------------------------------------------
// K8: logits = remove_diag-packed Gram of O via WMMA, scattered straight into
//     the [N2 x (N2-1)] output (output-write bound; ob is L2-resident).
// ---------------------------------------------------------------------------
__global__ __launch_bounds__(256) void k_logits(const __bf16* __restrict__ ob,
                                                float* __restrict__ out_logits) {
    const int lane = threadIdx.x & 31;
    const int wave = threadIdx.x >> 5;
    int tile = blockIdx.x * 8 + wave;          // 512*512 tiles
    int tm = tile >> 9;
    int tn = tile & 511;
    int m0 = tm << 4, n0 = tn << 4;
    const int half = lane >> 4;
    const int lrow = lane & 15;
    const __bf16* arow = ob + (size_t)(m0 + lrow) * DP;
    const __bf16* brow = ob + (size_t)(n0 + lrow) * DP;
    v8f acc = {};
#pragma unroll
    for (int kk = 0; kk < DP; kk += 32) {
        v16bf a = make_a_frag(arow + kk, half);
        v16bf b = make_b_frag(brow + kk, half);
        acc = __builtin_amdgcn_wmma_f32_16x16x32_bf16(
            false, a, false, b, (short)0, acc, false, false);
    }
    int mbase = m0 + half * 8;
    int g = n0 + lrow;                         // Gram column
#pragma unroll
    for (int v = 0; v < 8; ++v) {
        int r = mbase + v;                     // Gram row
        float val = acc[v];
        int outcol;
        bool keep = true;
        if (r < BSZ) {                         // top half: [l01 | rm_diag(l00)]
            if (g >= BSZ)       outcol = g - BSZ;
            else if (g < r)     outcol = BSZ + g;
            else if (g > r)     outcol = BSZ + g - 1;
            else                keep = false;
        } else {                               // bottom half: [l10 | rm_diag(l11)]
            int i = r - BSZ;
            if (g < BSZ)        outcol = g;
            else {
                int c = g - BSZ;
                if (c < i)      outcol = BSZ + c;
                else if (c > i) outcol = BSZ + c - 1;
                else            keep = false;
            }
        }
        if (keep) out_logits[(size_t)r * (N2 - 1) + outcol] = val;
    }
}

// ---------------------------------------------------------------------------
// K9: per-row LID(k=32) and LID(k=512).
//     dist row staged in LDS (32KB of the 320KB WGP pool), exact (k+1)-th
//     smallest via 32-step radix-select on the float bit pattern.
// ---------------------------------------------------------------------------
__global__ __launch_bounds__(256) void k_lid(const float* __restrict__ gram,
                                             const float* __restrict__ fen,
                                             float* __restrict__ out32,
                                             float* __restrict__ out512) {
    __shared__ float row[N2];
    __shared__ float sf[256];
    __shared__ int   si[256];
    int r = blockIdx.x, t = threadIdx.x;
    float nr = fen[r];
    float lmin = 3.4e38f;
    for (int i = t; i < N2; i += 256) {
        float d2 = nr + fen[i] - 2.0f * gram[(size_t)r * N2 + i];
        float d = sqrtf(fmaxf(d2, 0.0f));
        row[i] = d;
        lmin = fminf(lmin, d);
    }
    sf[t] = lmin; __syncthreads();
    for (int s = 128; s > 0; s >>= 1) {
        if (t < s) sf[t] = fminf(sf[t], sf[t + s]);
        __syncthreads();
    }
    float xmin = sf[0]; __syncthreads();       // self-distance (rank 0)

#pragma unroll
    for (int pass = 0; pass < 2; ++pass) {
        const int k = (pass == 0) ? 32 : 512;
        unsigned pfx = 0;
        int target = k;                        // 0-based rank k == d_k
        for (int b = 31; b >= 0; --b) {
            unsigned want = pfx >> b;          // candidate with bit b == 0
            int c = 0;
            for (int i = t; i < N2; i += 256)
                c += ((__float_as_uint(row[i]) >> b) == want) ? 1 : 0;
            si[t] = c; __syncthreads();
            for (int s = 128; s > 0; s >>= 1) {
                if (t < s) si[t] += si[t + s];
                __syncthreads();
            }
            int ctot = si[0]; __syncthreads();
            if (target >= ctot) { target -= ctot; pfx |= (1u << b); }
        }
        float vk = __uint_as_float(pfx);       // d_k
        float ls = 0.0f; int m = 0;
        for (int i = t; i < N2; i += 256) {
            float x = row[i];
            if (x < vk) { ++m; ls += logf(x / vk + EPSF); }
        }
        sf[t] = ls; si[t] = m; __syncthreads();
        for (int s = 128; s > 0; s >>= 1) {
            if (t < s) { sf[t] += sf[t + s]; si[t] += si[t + s]; }
            __syncthreads();
        }
        if (t == 0) {
            float total = sf[0];
            if (xmin < vk) total -= logf(xmin / vk + EPSF); // drop self term
            float lid = -(float)k / total;     // ties at vk: log(1+eps) ~= 0
            if (pass == 0) out32[r] = lid; else out512[r] = lid;
        }
        __syncthreads();
    }
}

// ---------------------------------------------------------------------------
// Host-side launcher
// ---------------------------------------------------------------------------
extern "C" void kernel_launch(void* const* d_in, const int* in_sizes, int n_in,
                              void* d_out, int out_size, void* d_ws, size_t ws_size,
                              hipStream_t stream) {
    (void)in_sizes; (void)n_in; (void)out_size; (void)ws_size;

    const float* fe0  = (const float*)d_in[0];
    const float* fe1  = (const float*)d_in[1];
    const float* p0   = (const float*)d_in[2];
    const float* p1   = (const float*)d_in[3];
    const float* z0   = (const float*)d_in[4];
    const float* z1   = (const float*)d_in[5];
    const float* cls0 = (const float*)d_in[6];
    const float* cls1 = (const float*)d_in[7];
    const int*   lab  = (const int*)  d_in[8];

    // Output layout (flat fp32, reference return order):
    float* out       = (float*)d_out;
    float* o_main    = out;                                    // [1]
    float* o_online  = out + 1;                                // [1]
    float* o_logits  = out + 2;                                // [N2 * (N2-1)]
    float* o_labels  = o_logits + (size_t)N2 * (N2 - 1);       // [N2]
    float* o_lid32   = o_labels + N2;                          // [N2]
    float* o_lid512  = o_lid32 + N2;                           // [N2]

    // Workspace carve-out (256B aligned slabs)
    char* w = (char*)d_ws;
    size_t off = 0;
    auto carve = [&](size_t bytes) -> void* {
        void* p = w + off;
        off = (off + bytes + 255) & ~(size_t)255;
        return p;
    };
    __bf16* feb     = (__bf16*)carve((size_t)N2 * DFE * sizeof(__bf16)); // 33.5MB
    float*  fen     = (float*) carve((size_t)N2 * sizeof(float));
    __bf16* ob      = (__bf16*)carve((size_t)N2 * DP * sizeof(__bf16));  // 4MB
    float*  pbyol   = (float*) carve((size_t)BSZ * sizeof(float));
    float*  pce     = (float*) carve((size_t)N2 * sizeof(float));
    float*  gram    = (float*) carve((size_t)N2 * N2 * sizeof(float));   // 268MB

    // --- small losses / bookkeeping ---
    k_prep_fe<<<N2, 256, 0, stream>>>(fe0, fe1, feb, fen);
    k_prep_p <<<N2, 256, 0, stream>>>(p0, p1, ob);
    k_byol   <<<BSZ, 256, 0, stream>>>(p0, p1, z0, z1, pbyol);
    k_ce     <<<N2, 256, 0, stream>>>(cls0, cls1, lab, pce);
    k_reduce_mean<<<1, 256, 0, stream>>>(pbyol, BSZ, 1.0f / BSZ, o_main);
    k_reduce_mean<<<1, 256, 0, stream>>>(pce, N2, 1.0f / N2, o_online);
    k_labels <<<N2 / 256, 256, 0, stream>>>(o_labels);

    // --- WMMA heavy lifting ---
    k_gram_fe<<<(N2 / GMT) * (N2 / GNT), 256, 0, stream>>>(feb, gram);
    k_lid    <<<N2, 256, 0, stream>>>(gram, fen, o_lid32, o_lid512);
    k_logits <<<(512 * 512) / 8, 256, 0, stream>>>(ob, o_logits);
}
